// LSTM_56401510531140
// MI455X (gfx1250) — matrix operand order repaired and checked by reading, not compiled, not run on hardware
//
#include <hip/hip_runtime.h>
#include <hip/hip_bf16.h>
#include <stdint.h>

// ---------------------------------------------------------------------------
// LSTM on MI455X (gfx1250):
//   Phase 0: convert x / W slices to bf16 (weights transposed, k-fast)
//   Phase 1: Zx = Xbf @ WxT via v_wmma_f32_16x16x32_bf16, 32x32 tile/wave
//   Phase 2: persistent recurrence, 8192 threads (4 k-split lanes per hidden
//            unit, shfl_xor reduce), bf16 weights streamed from L2,
//            h broadcast staged into LDS via TDM tensor_load_to_lds,
//            grid-wide monotonic barrier per step
//   Phase 3: out = h_last @ W_out + b_out
// ---------------------------------------------------------------------------

typedef __attribute__((ext_vector_type(16))) __bf16 v16bf;
typedef __attribute__((ext_vector_type(8)))  float  v8f;
typedef __attribute__((ext_vector_type(4)))  unsigned int v4ui;
typedef __attribute__((ext_vector_type(8)))  int    v8si;
typedef __attribute__((ext_vector_type(4)))  int    v4si;
typedef unsigned short u16;
typedef unsigned int   u32;

#define SEQ    2048
#define D_IN   1024
#define D_HID  2048
#define D_OUT  512
#define NGATE  4
#define NCOL   (NGATE * D_HID)      // 8192 gate columns

// ---- workspace layout (bytes) ----
#define WS_XB    ((size_t)0)                                  // [SEQ][D_IN] u16      4 MB
#define WS_WXT   (WS_XB  + (size_t)SEQ  * D_IN  * 2)          // [NCOL][D_IN] u16    16 MB
#define WS_WHT   (WS_WXT + (size_t)NCOL * D_IN  * 2)          // [NCOL][D_HID] u16   32 MB
#define WS_ZX    (WS_WHT + (size_t)NCOL * D_HID * 2)          // [SEQ][NCOL] f32     64 MB
#define WS_H     (WS_ZX  + (size_t)SEQ  * NCOL  * 4)          // [D_HID] f32
#define WS_BAR   (WS_H   + (size_t)D_HID * 4)                 // 1 u32

__device__ __forceinline__ u16 f2bf(float f) {
    u32 u = __float_as_uint(f);
    u32 r = 0x7fffu + ((u >> 16) & 1u);          // round-to-nearest-even
    return (u16)((u + r) >> 16);
}
__device__ __forceinline__ float bflo(u32 u) { return __uint_as_float(u << 16); }
__device__ __forceinline__ float bfhi(u32 u) { return __uint_as_float(u & 0xffff0000u); }
__device__ __forceinline__ float sigm(float x) { return 1.0f / (1.0f + __expf(-x)); }

// ---------------------------------------------------------------------------
// Phase 0a: x -> bf16  (layout unchanged: [t][k], k-fast)
__global__ void conv_x_kernel(const float* __restrict__ x, u16* __restrict__ xb, int n) {
    int i = blockIdx.x * blockDim.x + threadIdx.x;
    if (i < n) xb[i] = f2bf(x[i]);
}

// Phase 0b: one gate's W (3072 x 2048 row-major, [k][j]) ->
//   WxT[j][k] bf16 (k<1024)  and  WhT[j][k] bf16 (k<2048, rows 1024..3071)
__global__ void conv_w_kernel(const float* __restrict__ W,
                              u16* __restrict__ wxT_g, u16* __restrict__ whT_g) {
    int idx = blockIdx.x * blockDim.x + threadIdx.x;     // over D_HID * (D_IN + D_HID)
    if (idx >= D_HID * (D_IN + D_HID)) return;
    int j = idx / (D_IN + D_HID);
    int k = idx % (D_IN + D_HID);
    float w = W[(size_t)k * D_HID + j];
    if (k < D_IN) wxT_g[(size_t)j * D_IN + k]          = f2bf(w);
    else          whT_g[(size_t)j * D_HID + (k - D_IN)] = f2bf(w);
}

// ---------------------------------------------------------------------------
// Phase 1: Zx[t][n] = sum_k xb[t][k] * WxT[n][k]   (K = 1024, bf16 -> f32)
// 32x32 tile per wave: 2 A-frags x 2 B-frags -> 4 accumulators.
// 8 waves/block cover a 32x256 block tile (same A rows -> L0 reuse).
union FragBF { v16bf v; uint4 q[2]; };

__global__ __launch_bounds__(256) void gemm_zx_kernel(const u16* __restrict__ xb,
                                                      const u16* __restrict__ wxT,
                                                      float* __restrict__ zx) {
    const int wave = threadIdx.x >> 5;
    const int lane = threadIdx.x & 31;
    const int tile = blockIdx.x * 8 + wave;              // 16384 tiles (64 x 256)
    const int tt   = tile >> 8;                          // t-tile: 0..63
    const int tn   = tile & 255;                         // n-tile: 0..255
    const int khalf = lane >> 4;                         // 0 or 1
    const int l15   = lane & 15;
    const int m0 = tt * 32, n0 = tn * 32;

    // A fragment: lane row m, K runs [khalf*8 .. +7] and [16+khalf*8 .. +7]
    const u16* arow0 = xb  + (size_t)(m0 + l15)      * D_IN + khalf * 8;
    const u16* arow1 = xb  + (size_t)(m0 + 16 + l15) * D_IN + khalf * 8;
    // B fragment: lane col n, K runs [khalf*8 .. +7] and [16+khalf*8 .. +7]
    const u16* bcol0 = wxT + (size_t)(n0 + l15)      * D_IN + khalf * 8;
    const u16* bcol1 = wxT + (size_t)(n0 + 16 + l15) * D_IN + khalf * 8;

    v8f acc00 = {}, acc01 = {}, acc10 = {}, acc11 = {};
    for (int kk = 0; kk < D_IN; kk += 32) {
        FragBF a0, a1, b0, b1;
        a0.q[0] = *(const uint4*)(arow0 + kk);
        a0.q[1] = *(const uint4*)(arow0 + kk + 16);
        a1.q[0] = *(const uint4*)(arow1 + kk);
        a1.q[1] = *(const uint4*)(arow1 + kk + 16);
        b0.q[0] = *(const uint4*)(bcol0 + kk);
        b0.q[1] = *(const uint4*)(bcol0 + kk + 16);
        b1.q[0] = *(const uint4*)(bcol1 + kk);
        b1.q[1] = *(const uint4*)(bcol1 + kk + 16);
        acc00 = __builtin_amdgcn_wmma_f32_16x16x32_bf16(false, a0.v, false, b0.v, (short)0, acc00, false, false);
        acc01 = __builtin_amdgcn_wmma_f32_16x16x32_bf16(false, a0.v, false, b1.v, (short)0, acc01, false, false);
        acc10 = __builtin_amdgcn_wmma_f32_16x16x32_bf16(false, a1.v, false, b0.v, (short)0, acc10, false, false);
        acc11 = __builtin_amdgcn_wmma_f32_16x16x32_bf16(false, a1.v, false, b1.v, (short)0, acc11, false, false);
    }
    // C layout: VGPR r -> row (r + khalf*8), col = lane&15
    const int col0 = n0 + l15, col1 = col0 + 16;
    const int r0 = m0 + khalf * 8, r1 = r0 + 16;
#pragma unroll
    for (int r = 0; r < 8; ++r) {
        zx[(size_t)(r0 + r) * NCOL + col0] = acc00[r];
        zx[(size_t)(r0 + r) * NCOL + col1] = acc01[r];
        zx[(size_t)(r1 + r) * NCOL + col0] = acc10[r];
        zx[(size_t)(r1 + r) * NCOL + col1] = acc11[r];
    }
}

// ---------------------------------------------------------------------------
// Phase 2: persistent recurrence.  64 WGs x 128 threads = 8192 threads;
// 4 consecutive lanes share hidden unit j (k-range split 4 ways, shfl reduce).
__global__ void init_kernel(float* h_buf, u32* bar) {
    int i = blockIdx.x * blockDim.x + threadIdx.x;
    if (i < D_HID) h_buf[i] = 0.0f;
    if (i == 0) *bar = 0u;
}

__device__ __forceinline__ void grid_sync(u32* bar, int step, int nwg) {
    __syncthreads();
    if (threadIdx.x == 0) {
        __builtin_amdgcn_fence(__ATOMIC_RELEASE, "agent");   // flush h store
        __hip_atomic_fetch_add(bar, 1u, __ATOMIC_RELAXED, __HIP_MEMORY_SCOPE_AGENT);
        const u32 target = (u32)(step + 1) * (u32)nwg;       // monotonic, no reset race
        while (__hip_atomic_load(bar, __ATOMIC_RELAXED, __HIP_MEMORY_SCOPE_AGENT) < target)
            __builtin_amdgcn_s_sleep(1);
        __builtin_amdgcn_fence(__ATOMIC_ACQUIRE, "agent");   // invalidate for next h load
    }
    __syncthreads();
}

// Stage h_buf[0..2047] (8KB) into LDS. Preferred path: one TDM descriptor
// (2D, 1 row x 2048 elems, data_size=4B) issued by wave 0; fallback: lane loads.
__device__ __forceinline__ void stage_h(const float* h_buf, float* hs) {
#if __has_builtin(__builtin_amdgcn_tensor_load_to_lds) && __has_builtin(__builtin_amdgcn_s_wait_tensorcnt)
    if (threadIdx.x < 32) {                     // single wave issues the DMA
        const unsigned long long ga = (unsigned long long)(uintptr_t)h_buf;
        const u32 lds_off = (u32)(uintptr_t)hs; // LDS aperture: low 32 bits = LDS byte addr
        v4ui g0; v8si g1; v4si z4 = {0, 0, 0, 0};
        // D# group 0: count=1, lds_addr, global_addr[56:0], type=2 (ISA 8.3)
        g0[0] = 1u;
        g0[1] = lds_off;
        g0[2] = (u32)(ga & 0xffffffffu);
        g0[3] = (u32)((ga >> 32) & 0x01ffffffu) | (2u << 30);
        // D# group 1 (ISA 8.4): data_size=2 (4B), tensor_dim0=2048, tensor_dim1=1,
        // tile_dim0=2048, tile_dim1=1, tensor_dim0_stride=2048
        g1[0] = (int)(2u << 16);                // wg_mask=0 | data_size=4B
        g1[1] = (int)((u32)D_HID << 16);        // tensor_dim0[15:0] @ bits 63:48
        g1[2] = (int)(1u << 16);                // tensor_dim0 hi=0 | tensor_dim1 lo=1
        g1[3] = (int)((u32)D_HID << 16);        // tensor_dim1 hi=0 | tile_dim0=2048
        g1[4] = 1;                              // tile_dim1=1 | tile_dim2=0
        g1[5] = D_HID;                          // tensor_dim0_stride lo32
        g1[6] = 0; g1[7] = 0;
#if __clang_major__ >= 23
        v8si z8 = {0, 0, 0, 0, 0, 0, 0, 0};
        __builtin_amdgcn_tensor_load_to_lds(g0, g1, z4, z4, z8, 0);
#else
        __builtin_amdgcn_tensor_load_to_lds(g0, g1, z4, z4, 0);
#endif
        __builtin_amdgcn_s_wait_tensorcnt(0);
    }
#else
    for (int i = threadIdx.x; i < D_HID; i += blockDim.x) hs[i] = h_buf[i];
#endif
}

__global__ __launch_bounds__(128, 1) void lstm_recur_kernel(
        const u16*  __restrict__ whT,   // [NCOL][D_HID] bf16, k-fast
        const float* __restrict__ zx,   // [SEQ][NCOL]
        const float* __restrict__ b_f, const float* __restrict__ b_i,
        const float* __restrict__ b_c, const float* __restrict__ b_o,
        float* h_buf, u32* bar) {
    __shared__ __align__(16) float hs[D_HID];
    const int tid = blockIdx.x * blockDim.x + threadIdx.x;  // 0..8191
    const int j   = tid >> 2;                               // hidden unit 0..2047
    const int q   = tid & 3;                                // k-quarter
    const int k0  = q * (D_HID / 4);                        // 0,512,1024,1536

    const u16* rowF = whT + (size_t)(0 * D_HID + j) * D_HID + k0;
    const u16* rowI = whT + (size_t)(1 * D_HID + j) * D_HID + k0;
    const u16* rowG = whT + (size_t)(2 * D_HID + j) * D_HID + k0;
    const u16* rowO = whT + (size_t)(3 * D_HID + j) * D_HID + k0;
    const float bF = b_f[j], bI = b_i[j], bC = b_c[j], bO = b_o[j];

    float c = 0.0f;
    for (int t = 0; t < SEQ; ++t) {
        __syncthreads();                 // protect hs from previous-step readers
        stage_h(h_buf, hs);
        __syncthreads();

        float aF = 0.f, aI = 0.f, aG = 0.f, aO = 0.f;
        for (int k = 0; k < D_HID / 4; k += 8) {
            __builtin_prefetch(rowF + k + 256, 0, 0);        // global_prefetch_b8
            const uint4 qF = *(const uint4*)(rowF + k);
            const uint4 qI = *(const uint4*)(rowI + k);
            const uint4 qG = *(const uint4*)(rowG + k);
            const uint4 qO = *(const uint4*)(rowO + k);
#pragma unroll
            for (int e = 0; e < 4; ++e) {
                const u32 uF = (&qF.x)[e], uI = (&qI.x)[e];
                const u32 uG = (&qG.x)[e], uO = (&qO.x)[e];
                const float h0 = hs[k0 + k + 2 * e], h1 = hs[k0 + k + 2 * e + 1];
                aF = fmaf(bflo(uF), h0, aF); aF = fmaf(bfhi(uF), h1, aF);
                aI = fmaf(bflo(uI), h0, aI); aI = fmaf(bfhi(uI), h1, aI);
                aG = fmaf(bflo(uG), h0, aG); aG = fmaf(bfhi(uG), h1, aG);
                aO = fmaf(bflo(uO), h0, aO); aO = fmaf(bfhi(uO), h1, aO);
            }
        }
        // butterfly reduce across the 4 k-split lanes (intra-wave, wave32)
        aF += __shfl_xor(aF, 1, 32); aF += __shfl_xor(aF, 2, 32);
        aI += __shfl_xor(aI, 1, 32); aI += __shfl_xor(aI, 2, 32);
        aG += __shfl_xor(aG, 1, 32); aG += __shfl_xor(aG, 2, 32);
        aO += __shfl_xor(aO, 1, 32); aO += __shfl_xor(aO, 2, 32);

        const float* z = zx + (size_t)t * NCOL;
        const float f  = sigm(z[0 * D_HID + j] + bF + aF);
        const float ii = sigm(z[1 * D_HID + j] + bI + aI);
        const float g  = tanhf(z[2 * D_HID + j] + bC + aG);
        const float o  = sigm(z[3 * D_HID + j] + bO + aO);
        c = c * f + ii * g;              // identical in all 4 lanes (deterministic)
        if (q == 0) h_buf[j] = tanhf(c) * o;

        grid_sync(bar, t, (int)gridDim.x);
    }
}

// ---------------------------------------------------------------------------
// Phase 3: out[o] = b_out[o] + sum_k h[k] * W_out[k][o]
__global__ void out_gemv_kernel(const float* __restrict__ h,
                                const float* __restrict__ Wout,
                                const float* __restrict__ bout,
                                float* __restrict__ out) {
    int o = blockIdx.x * blockDim.x + threadIdx.x;
    if (o >= D_OUT) return;
    float acc = bout[o];
    for (int k = 0; k < D_HID; ++k)
        acc = fmaf(h[k], Wout[(size_t)k * D_OUT + o], acc);
    out[o] = acc;
}

// ---------------------------------------------------------------------------
extern "C" void kernel_launch(void* const* d_in, const int* in_sizes, int n_in,
                              void* d_out, int out_size, void* d_ws, size_t ws_size,
                              hipStream_t stream) {
    const float* x    = (const float*)d_in[0];
    const float* W_f  = (const float*)d_in[1];
    const float* b_f  = (const float*)d_in[2];
    const float* W_i  = (const float*)d_in[3];
    const float* b_i  = (const float*)d_in[4];
    const float* W_c  = (const float*)d_in[5];
    const float* b_c  = (const float*)d_in[6];
    const float* W_o  = (const float*)d_in[7];
    const float* b_o  = (const float*)d_in[8];
    const float* Wout = (const float*)d_in[9];
    const float* bout = (const float*)d_in[10];
    float* out = (float*)d_out;

    char* ws = (char*)d_ws;
    u16*   xb  = (u16*)(ws + WS_XB);
    u16*   wxT = (u16*)(ws + WS_WXT);
    u16*   whT = (u16*)(ws + WS_WHT);
    float* zx  = (float*)(ws + WS_ZX);
    float* hb  = (float*)(ws + WS_H);
    u32*   bar = (u32*)(ws + WS_BAR);

    // Phase 0: bf16 conversions
    {
        const int n = SEQ * D_IN;
        conv_x_kernel<<<(n + 255) / 256, 256, 0, stream>>>(x, xb, n);
        const int nw = D_HID * (D_IN + D_HID);
        const int gb = (nw + 255) / 256;
        const float* Ws[NGATE] = {W_f, W_i, W_c, W_o};
        for (int g = 0; g < NGATE; ++g)
            conv_w_kernel<<<gb, 256, 0, stream>>>(
                Ws[g],
                wxT + (size_t)g * D_HID * D_IN,
                whT + (size_t)g * D_HID * D_HID);
    }

    // Phase 1: Zx via WMMA bf16 (16384 tiles of 32x32, 8 waves/block)
    gemm_zx_kernel<<<16384 / 8, 256, 0, stream>>>(xb, wxT, zx);

    // Phase 2: persistent recurrence (64 WGs x 128 threads, all co-resident)
    init_kernel<<<(D_HID + 255) / 256, 256, 0, stream>>>(hb, bar);
    lstm_recur_kernel<<<64, 128, 0, stream>>>(whT, zx, b_f, b_i, b_c, b_o, hb, bar);

    // Phase 3: output projection
    out_gemv_kernel<<<(D_OUT + 255) / 256, 256, 0, stream>>>(hb, Wout, bout, out);
}
